// AWQWeight_30786325578069
// MI455X (gfx1250) — matrix operand order repaired and checked by reading, not compiled, not run on hardware
//
#include <hip/hip_runtime.h>
#include <stdint.h>

typedef __attribute__((ext_vector_type(16))) _Float16 v16h;
typedef __attribute__((ext_vector_type(8)))  _Float16 v8h;
typedef __attribute__((ext_vector_type(4)))  _Float16 v4h;
typedef __attribute__((ext_vector_type(8)))  float    v8f;
typedef __attribute__((ext_vector_type(8)))  int      v8i;

#define B_DIM 8
#define K_DIM 4096
#define N_DIM 11008
#define GSZ   128
#define NB8   1376                 // N/8 packed dwords per qweight row
#define COLS_PER_BLOCK 256         // 8 waves * 32 cols
#define PARTS 8                    // K split for parallelism
#define NGROUPS 32                 // K / GSZ
#define XPITCH 136                 // f16 pitch of LDS x tile (bank spread)
#define QBUF   (GSZ * 32)          // dwords per qweight LDS buffer
#define XBUF   (16 * XPITCH)       // halves per x LDS buffer

// Per-group row sums of x: S[g][b] = sum_{k in group g} x[b][k]
__device__ float g_xsums[NGROUPS * B_DIM];

__global__ __launch_bounds__(256)
void xsums_kernel(const float* __restrict__ x) {
    const int t = threadIdx.x;       // 256 threads = 32 groups * 8 rows
    const int g = t >> 3, b = t & 7;
    const float* xp = x + (size_t)b * K_DIM + g * GSZ;
    float s = 0.f;
    #pragma unroll 8
    for (int i = 0; i < GSZ; ++i) s += xp[i];
    g_xsums[g * B_DIM + b] = s;
}

struct H16 { v8h lo, hi; };

// Per group g: D' = x_f16 @ f16(0x6400|nib) via WMMA from zero, then
// accF += s*D' - s*(1024+z)*S[row].  1024+iw is exact in fp16.
__global__ __launch_bounds__(256)
void awq_wmma_kernel(const float* __restrict__ x,
                     const int*   __restrict__ qweight,
                     const float* __restrict__ scales,
                     const int*   __restrict__ qzeros,
                     const float* __restrict__ bias,
                     float* __restrict__ dst_base,
                     int parts, int add_bias)
{
    __shared__ int      lds_q[2 * QBUF];          // 2 x 16 KB qweight chunks
    __shared__ _Float16 x_lds[2 * XBUF];          // 2 x f16 x tiles (rows 8..15 zero)

    const int tid  = threadIdx.x;
    const int lane = tid & 31;
    const int wave = tid >> 5;
    const int l16  = lane & 15;
    const int half = lane >> 4;

    const int nblock = blockIdx.x * COLS_PER_BLOCK;
    const int nw     = nblock + wave * 32;
    const int nb0    = nblock >> 3;

    // AWQ inverse nibble order: logical col c lives at nibble (c%4)*2 + c/4
    const int c8    = l16 & 7;
    const int shift = 4 * ((c8 & 3) * 2 + (c8 >> 2));

    const int chunks_per_part = NGROUPS / parts;
    const int g0 = blockIdx.y * chunks_per_part;

    float* dst = dst_base + (size_t)blockIdx.y * (parts > 1 ? (size_t)B_DIM * N_DIM : 0);

    const int n_s0 = nw + l16;
    const int n_s1 = nw + 16 + l16;

    // ---- async cooperative-load geometry: b128 per thread, 4 steps/chunk ----
    const int dl4   = (tid & 7) * 4;              // dword-local base (multiple of 4)
    const int rbase = tid >> 3;                   // row 0..31 within a 32-row step
    const unsigned ldq_base = (unsigned)(uintptr_t)&lds_q[rbase * 32 + dl4];
    unsigned voff = (unsigned)((unsigned)(g0 * GSZ + rbase) * (NB8 * 4u)
                               + (unsigned)(nb0 + dl4) * 4u);

    // x-fill geometry: thread -> (row, 4 consecutive k)
    const int xrow = tid >> 5;                    // 0..7
    const int xkof = (tid & 31) * 4;              // 0..124
    const float* xg = x + (size_t)xrow * K_DIM + g0 * GSZ + xkof;

    // zero the padding rows (8..15) of both f16 x tiles once
    #pragma unroll
    for (int b = 0; b < 2; ++b) {
        unsigned* zp = (unsigned*)&x_lds[b * XBUF + 8 * XPITCH];
        for (int i = tid; i < 8 * XPITCH / 2; i += 256) zp[i] = 0u;
    }

    // per-strip group-constant pointers (bumped by constant strides)
    const int*   qzp = qzeros + (size_t)g0 * NB8 + (n_s0 >> 3);
    const float* scp = scales + (size_t)g0 * N_DIM + n_s0;
    const float* sp  = g_xsums + g0 * B_DIM;      // uniform -> scalar loads

    const float cmask = (half == 0) ? 1.f : 0.f;  // pad rows: no correction term

    float accF0[8] = {0.f, 0.f, 0.f, 0.f, 0.f, 0.f, 0.f, 0.f};
    float accF1[8] = {0.f, 0.f, 0.f, 0.f, 0.f, 0.f, 0.f, 0.f};

    // ---- prologue: start chunk 0 into buffer 0 ----
    #pragma unroll
    for (int i = 0; i < 4; ++i) {
        asm volatile("global_load_async_to_lds_b128 %0, %1, %2"
                     :: "v"(ldq_base + (unsigned)i * (32u * 32u * 4u)),
                        "v"(voff + (unsigned)i * (32u * NB8 * 4u)),
                        "s"(qweight)
                     : "memory");
    }
    voff += GSZ * NB8 * 4u;
    {
        const float4 xf = *(const float4*)xg;
        v4h h = { (_Float16)xf.x, (_Float16)xf.y, (_Float16)xf.z, (_Float16)xf.w };
        *(v4h*)&x_lds[xrow * XPITCH + xkof] = h;
        xg += GSZ;
    }

    for (int gi = 0; gi < chunks_per_part; ++gi) {
        const int bsel  = gi & 1;
        const int bnext = bsel ^ 1;

        // loads for chunk gi are the only outstanding asyncs -> drain, then rendezvous
        asm volatile("s_wait_asynccnt 0x0" ::: "memory");
        __syncthreads();   // everyone's chunk-gi data ready; buffers from gi-1 now free

        // ---- kick off chunk gi+1 into the other buffer (overlaps compute) ----
        if (gi + 1 < chunks_per_part) {
            const unsigned ldq_n = ldq_base + (unsigned)bnext * (QBUF * 4u);
            #pragma unroll
            for (int i = 0; i < 4; ++i) {
                asm volatile("global_load_async_to_lds_b128 %0, %1, %2"
                             :: "v"(ldq_n + (unsigned)i * (32u * 32u * 4u)),
                                "v"(voff + (unsigned)i * (32u * NB8 * 4u)),
                                "s"(qweight)
                             : "memory");
            }
            voff += GSZ * NB8 * 4u;
            const float4 xf = *(const float4*)xg;
            v4h h = { (_Float16)xf.x, (_Float16)xf.y, (_Float16)xf.z, (_Float16)xf.w };
            *(v4h*)&x_lds[bnext * XBUF + xrow * XPITCH + xkof] = h;
            xg += GSZ;
        }

        // group constants for chunk gi
        const int   z0  = (int)(((unsigned)qzp[0] >> shift) & 0xFu);
        const int   z1  = (int)(((unsigned)qzp[2] >> shift) & 0xFu);
        const float sc0 = scp[0];
        const float sc1 = scp[16];
        float u[8];
        #pragma unroll
        for (int d = 0; d < 8; ++d) u[d] = sp[d];
        qzp += NB8;
        scp += N_DIM;
        sp  += B_DIM;

        const int        qoff = bsel * QBUF;
        const _Float16*  xl   = &x_lds[bsel * XBUF + l16 * XPITCH + half * 16];

        v8f D0 = {0.f, 0.f, 0.f, 0.f, 0.f, 0.f, 0.f, 0.f};
        v8f D1 = {0.f, 0.f, 0.f, 0.f, 0.f, 0.f, 0.f, 0.f};

        #pragma unroll
        for (int kb = 0; kb < 4; ++kb) {
            // A fragment: two b128 LDS reads, zero VALU
            H16 t;
            t.lo = *(const v8h*)(xl + kb * 32);
            t.hi = *(const v8h*)(xl + kb * 32 + 8);
            const v16h A = __builtin_bit_cast(v16h, t);

            // B fragments: pure bit-twiddle nibble -> fp16 (1024 + iw)
            v8i P0, P1;
            const int dl0 = 4 * wave + (l16 >> 3);
            const int dl1 = dl0 + 2;
            #pragma unroll
            for (int v = 0; v < 8; ++v) {
                const int kl = kb * 32 + half * 16 + 2 * v;
                const unsigned q00 = (unsigned)lds_q[qoff + kl * 32 + dl0];
                const unsigned q01 = (unsigned)lds_q[qoff + (kl + 1) * 32 + dl0];
                const unsigned q10 = (unsigned)lds_q[qoff + kl * 32 + dl1];
                const unsigned q11 = (unsigned)lds_q[qoff + (kl + 1) * 32 + dl1];
                const unsigned lo0 = (q00 >> shift) & 0xFu;
                const unsigned hi0 = (q01 >> shift) & 0xFu;
                const unsigned lo1 = (q10 >> shift) & 0xFu;
                const unsigned hi1 = (q11 >> shift) & 0xFu;
                P0[v] = (int)((hi0 << 16) | lo0 | 0x64006400u);
                P1[v] = (int)((hi1 << 16) | lo1 | 0x64006400u);
            }
            const v16h B0 = __builtin_bit_cast(v16h, P0);
            const v16h B1 = __builtin_bit_cast(v16h, P1);

            D0 = __builtin_amdgcn_wmma_f32_16x16x32_f16(
                false, A, false, B0, (short)0, D0, false, false);
            D1 = __builtin_amdgcn_wmma_f32_16x16x32_f16(
                false, A, false, B1, (short)0, D1, false, false);
        }

        // fold group: accF += s*D' - s*(1024+z)*S[row]  (pad rows masked via cmask)
        const float c0 = sc0 * (1024.f + (float)z0) * cmask;
        const float c1 = sc1 * (1024.f + (float)z1) * cmask;
        #pragma unroll
        for (int d = 0; d < 8; ++d) {
            accF0[d] = __builtin_fmaf(sc0, D0[d], __builtin_fmaf(-c0, u[d], accF0[d]));
            accF1[d] = __builtin_fmaf(sc1, D1[d], __builtin_fmaf(-c1, u[d], accF1[d]));
        }
    }

    // D layout: lanes 0-15 hold rows 0..7 (real batch), col = lane
    if (half == 0) {
        const float b0v = add_bias ? bias[n_s0] : 0.f;
        const float b1v = add_bias ? bias[n_s1] : 0.f;
        #pragma unroll
        for (int d = 0; d < 8; ++d) {
            dst[(size_t)d * N_DIM + n_s0] = accF0[d] + b0v;
            dst[(size_t)d * N_DIM + n_s1] = accF1[d] + b1v;
        }
    }
}

// Fixed-order reduction over K-parts (deterministic, no atomics) + bias add
__global__ __launch_bounds__(256)
void awq_reduce_kernel(const float* __restrict__ ws,
                       const float* __restrict__ bias,
                       float* __restrict__ out, int parts)
{
    const int idx = blockIdx.x * blockDim.x + threadIdx.x;
    if (idx >= B_DIM * N_DIM) return;
    const int n = idx % N_DIM;
    float s = bias[n];
    for (int p = 0; p < parts; ++p)
        s += ws[(size_t)p * B_DIM * N_DIM + idx];
    out[idx] = s;
}

extern "C" void kernel_launch(void* const* d_in, const int* in_sizes, int n_in,
                              void* d_out, int out_size, void* d_ws, size_t ws_size,
                              hipStream_t stream) {
    const float* x    = (const float*)d_in[0];
    const int*   qw   = (const int*)d_in[1];
    const float* sc   = (const float*)d_in[2];
    const int*   qz   = (const int*)d_in[3];
    const float* bias = (const float*)d_in[4];
    float* out = (float*)d_out;

    xsums_kernel<<<1, 256, 0, stream>>>(x);

    const size_t need = (size_t)PARTS * B_DIM * N_DIM * sizeof(float);
    if (d_ws != nullptr && ws_size >= need) {
        float* ws = (float*)d_ws;
        dim3 grid(N_DIM / COLS_PER_BLOCK, PARTS);
        awq_wmma_kernel<<<grid, 256, 0, stream>>>(x, qw, sc, qz, bias, ws,
                                                  PARTS, /*add_bias=*/0);
        const int total = B_DIM * N_DIM;
        awq_reduce_kernel<<<(total + 255) / 256, 256, 0, stream>>>(ws, bias, out, PARTS);
    } else {
        dim3 grid(N_DIM / COLS_PER_BLOCK, 1);
        awq_wmma_kernel<<<grid, 256, 0, stream>>>(x, qw, sc, qz, bias, out,
                                                  1, /*add_bias=*/1);
    }
}
